// RNNDecoder_38835094290830
// MI455X (gfx1250) — compile-verified, matrix operand order checked by reading, not run on hardware
//
#include <hip/hip_runtime.h>
#include <hip/hip_bf16.h>
#include <math.h>

// ---------------------------------------------------------------------------
// RNN attention decoder for MI455X (gfx1250, wave32, WMMA f16 16x16x32).
// B=32, T=64, S=256, DW=DM=DENC=512, DK=64, POOL=2.
// GEMMs: v_wmma_f32_16x16x32_f16. Weights/embeddings pre-converted to f16.
// LDS staging: double-buffered global_load_async_to_lds_b64/b128 pipelined
// against WMMA; exact-depth s_wait_asynccnt for overlap.
// ---------------------------------------------------------------------------

typedef __attribute__((ext_vector_type(16))) _Float16 v16h;
typedef __attribute__((ext_vector_type(8)))  _Float16 v8h;
typedef __attribute__((ext_vector_type(4)))  _Float16 v4h;
typedef __attribute__((ext_vector_type(2)))  _Float16 v2h;
typedef __attribute__((ext_vector_type(8)))  float    v8f;

#define KP 40   // padded K-stride in halfs: 80B rows keep every chunk 8/16B aligned

// ---- CDNA5 async global->LDS copies (ASYNCcnt-tracked) --------------------
__device__ __forceinline__ void async_cp_b128(_Float16* dst, const _Float16* src) {
  unsigned lo = (unsigned)(unsigned long long)(void*)dst;  // low 32b of generic = LDS offset
  asm volatile("global_load_async_to_lds_b128 %0, %1, off"
               :: "v"(lo), "v"((unsigned long long)(const void*)src) : "memory");
}
__device__ __forceinline__ void async_cp_b64(_Float16* dst, const _Float16* src) {
  unsigned lo = (unsigned)(unsigned long long)(void*)dst;
  asm volatile("global_load_async_to_lds_b64 %0, %1, off"
               :: "v"(lo), "v"((unsigned long long)(const void*)src) : "memory");
}
template <int N> __device__ __forceinline__ void async_wait() {
  asm volatile("s_wait_asynccnt %0" :: "i"(N) : "memory");
}

// A tile: 32 rows x 32 halfs; every lane moves 8B -> 1 async op per wave.
__device__ __forceinline__ void stageA(_Float16* lds, const _Float16* g,
                                       size_t rowStride, int tid) {
  int r = tid >> 3, ch = (tid & 7) * 4;
  async_cp_b64(lds + r * KP + ch, g + (size_t)r * rowStride + ch);
}
// B tile: 64 rows x 32 halfs; every lane moves 16B -> 1 async op per wave.
__device__ __forceinline__ void stageB(_Float16* lds, const _Float16* g,
                                       size_t rowStride, int tid) {
  int r = tid >> 2, ch = (tid & 3) * 8;
  async_cp_b128(lds + r * KP + ch, g + (size_t)r * rowStride + ch);
}
// 3 gate B tiles (192 rows x 32 halfs) -> 3 async ops per wave.
__device__ __forceinline__ void stageB3(_Float16* lds, const _Float16* Wbase,
                                        size_t kdim, int tid) {
  #pragma unroll
  for (int i = tid; i < 768; i += 256) {
    int g = i >> 8, r = (i >> 2) & 63, ch = (i & 3) * 8;
    async_cp_b128(lds + (g * 64 + r) * KP + ch,
                  Wbase + ((size_t)g * 512 + r) * kdim + ch);
  }
}

// ---- WMMA fragment path ---------------------------------------------------
__device__ __forceinline__ v16h frag_ld(const _Float16* tile, int rsel, int half) {
  const _Float16* p = tile + rsel * KP + half * 8;
  v8h lo = *(const v8h*)(p);
  v8h hi = *(const v8h*)(p + 16);
  return __builtin_shufflevector(lo, hi, 0,1,2,3,4,5,6,7,8,9,10,11,12,13,14,15);
}
__device__ __forceinline__ v8f wmma_step(const _Float16* lA, const _Float16* lB,
                                         int mt, int nt, int lane, v8f acc) {
  int rsel = lane & 15, half = lane >> 4;
  v16h a = frag_ld(lA + mt * 16 * KP, rsel, half);
  v16h b = frag_ld(lB + nt * 16 * KP, rsel, half);
  return __builtin_amdgcn_wmma_f32_16x16x32_f16(false, a, false, b, (short)0, acc,
                                                false, false);
}
__device__ __forceinline__ float sigf(float x) { return 1.0f / (1.0f + expf(-x)); }

// ---------------------------------------------------------------------------
// One-time converters.
// ---------------------------------------------------------------------------
__global__ __launch_bounds__(256) void k_cvt(const float* __restrict__ src,
                                             _Float16* __restrict__ dst, int n) {
  int i = (blockIdx.x * 256 + threadIdx.x) * 8;
  if (i + 8 <= n) {
    float4 a = *(const float4*)(src + i);
    float4 b = *(const float4*)(src + i + 4);
    v8h o;
    o[0] = (_Float16)a.x; o[1] = (_Float16)a.y; o[2] = (_Float16)a.z; o[3] = (_Float16)a.w;
    o[4] = (_Float16)b.x; o[5] = (_Float16)b.y; o[6] = (_Float16)b.z; o[7] = (_Float16)b.w;
    *(v8h*)(dst + i) = o;
  }
}

__global__ __launch_bounds__(256) void k_emb(const int* __restrict__ tgt,
                                             const float* __restrict__ wemb,
                                             _Float16* __restrict__ emb16) {
  int row = blockIdx.x;  // b*64+t
  int tok = tgt[row];
  const float* srcp = wemb + (size_t)tok * 512;
  int c = threadIdx.x * 2;
  float2 a = *(const float2*)(srcp + c);
  v2h o; o[0] = (_Float16)a.x; o[1] = (_Float16)a.y;
  *(v2h*)(emb16 + (size_t)row * 512 + c) = o;
}

// ---------------------------------------------------------------------------
// precompute = enc_output @ W_pre^T + b_pre   [8192,512]x[64,512]^T (once)
// ---------------------------------------------------------------------------
__global__ __launch_bounds__(256) void k_pre(const float* __restrict__ enc,
                                             const _Float16* __restrict__ Wpre16,
                                             const float* __restrict__ b_pre,
                                             float* __restrict__ out) {
  __shared__ _Float16 lA[32 * KP];
  __shared__ _Float16 lB[64 * KP];
  const int tid = threadIdx.x, lane = tid & 31, w = tid >> 5;
  const int mt = w & 1, nt = w >> 1;
  const int m0 = blockIdx.x * 32;
  v8f acc = {0, 0, 0, 0, 0, 0, 0, 0};
  for (int kk = 0; kk < 512; kk += 32) {
    {  // A: inline f32->f16, 4 floats/thread
      int r = tid >> 3, ch = (tid & 7) * 4;
      float4 a4 = *(const float4*)(enc + (size_t)(m0 + r) * 512 + kk + ch);
      v4h o; o[0] = (_Float16)a4.x; o[1] = (_Float16)a4.y;
      o[2] = (_Float16)a4.z; o[3] = (_Float16)a4.w;
      *(v4h*)(lA + r * KP + ch) = o;
    }
    stageB(lB, Wpre16 + kk, 512, tid);
    async_wait<0>();
    __syncthreads();
    acc = wmma_step(lA, lB, mt, nt, lane, acc);
    __syncthreads();
  }
  const int col = nt * 16 + (lane & 15);
  const int half = lane >> 4;
  const float bp = b_pre[col];
  for (int v = 0; v < 8; ++v) {
    int m = m0 + mt * 16 + v + 8 * half;
    out[(size_t)m * 64 + col] = acc[v] + bp;
  }
}

// ---------------------------------------------------------------------------
// h0 = tanh(hidden @ W_init^T + b_init), double-buffered async pipeline.
// ---------------------------------------------------------------------------
__global__ __launch_bounds__(256) void k_init(const _Float16* __restrict__ hid16,
                                              const _Float16* __restrict__ Winit16,
                                              const float* __restrict__ b_init,
                                              float* __restrict__ h0,
                                              _Float16* __restrict__ h016) {
  __shared__ _Float16 lA[2][32 * KP];
  __shared__ _Float16 lB[2][64 * KP];
  const int tid = threadIdx.x, lane = tid & 31, w = tid >> 5;
  const int mt = w & 1, nt = w >> 1;
  const int c0 = blockIdx.x * 64;
  v8f acc = {0, 0, 0, 0, 0, 0, 0, 0};
  stageA(lA[0], hid16, 512, tid);
  stageB(lB[0], Winit16 + (size_t)c0 * 512, 512, tid);
  for (int c = 0; c < 16; ++c) {
    int cur = c & 1;
    if (c + 1 < 16) {
      int kk = (c + 1) * 32;
      stageA(lA[cur ^ 1], hid16 + kk, 512, tid);
      stageB(lB[cur ^ 1], Winit16 + (size_t)c0 * 512 + kk, 512, tid);
      async_wait<2>();
    } else {
      async_wait<0>();
    }
    __syncthreads();
    acc = wmma_step(lA[cur], lB[cur], mt, nt, lane, acc);
    __syncthreads();
  }
  const int col = c0 + nt * 16 + (lane & 15);
  const int half = lane >> 4;
  const float bi = b_init[col];
  for (int v = 0; v < 8; ++v) {
    int m = mt * 16 + v + 8 * half;
    float hv = tanhf(acc[v] + bi);
    h0[(size_t)m * 512 + col] = hv;
    h016[(size_t)m * 512 + col] = (_Float16)hv;
  }
}

// ---------------------------------------------------------------------------
// Fused GRU step. Unified 48-chunk pipeline: chunks 0..31 = input side
// ([emb|ctx] x W_ih, K=1024), 32..47 = hidden side (h x W_hh, K=512).
// 3 gate B-tiles staged per chunk -> 1 barrier pair per 3 WMMAs.
// ---------------------------------------------------------------------------
__global__ __launch_bounds__(256) void k_gru(const _Float16* __restrict__ emb16,
                                             const _Float16* __restrict__ ctx16,
                                             const _Float16* __restrict__ h16cur,
                                             const float* __restrict__ hcur,
                                             const _Float16* __restrict__ Wih16,
                                             const _Float16* __restrict__ Whh16,
                                             const float* __restrict__ b_ih,
                                             const float* __restrict__ b_hh,
                                             float* __restrict__ hnxt,
                                             _Float16* __restrict__ hnxt16, int t) {
  __shared__ _Float16 lA[2][32 * KP];
  __shared__ _Float16 lB[2][3 * 64 * KP];
  const int tid = threadIdx.x, lane = tid & 31, w = tid >> 5;
  const int mt = w & 1, nt = w >> 1;
  const int c0 = blockIdx.x * 64;
  v8f accI[3], accH[3];
  for (int g = 0; g < 3; ++g) {
    accI[g] = (v8f){0, 0, 0, 0, 0, 0, 0, 0};
    accH[g] = (v8f){0, 0, 0, 0, 0, 0, 0, 0};
  }

  auto stage_chunk = [&](int c, int buf) {
    if (c < 32) {
      int kk = c * 32;
      if (kk < 512) stageA(lA[buf], emb16 + (size_t)t * 512 + kk, 64 * 512, tid);
      else          stageA(lA[buf], ctx16 + (kk - 512), 512, tid);
      stageB3(lB[buf], Wih16 + (size_t)c0 * 1024 + kk, 1024, tid);
    } else {
      int kk = (c - 32) * 32;
      stageA(lA[buf], h16cur + kk, 512, tid);
      stageB3(lB[buf], Whh16 + (size_t)c0 * 512 + kk, 512, tid);
    }
  };

  stage_chunk(0, 0);
  for (int c = 0; c < 48; ++c) {
    int cur = c & 1;
    if (c + 1 < 48) { stage_chunk(c + 1, cur ^ 1); async_wait<4>(); }
    else            { async_wait<0>(); }
    __syncthreads();
    if (c < 32) {
      for (int g = 0; g < 3; ++g)
        accI[g] = wmma_step(lA[cur], lB[cur] + g * 64 * KP, mt, nt, lane, accI[g]);
    } else {
      for (int g = 0; g < 3; ++g)
        accH[g] = wmma_step(lA[cur], lB[cur] + g * 64 * KP, mt, nt, lane, accH[g]);
    }
    __syncthreads();
  }

  const int col = c0 + nt * 16 + (lane & 15);
  const int half = lane >> 4;
  const float bir = b_ih[col], biz = b_ih[512 + col], bin_ = b_ih[1024 + col];
  const float bhr = b_hh[col], bhz = b_hh[512 + col], bhn = b_hh[1024 + col];
  for (int v = 0; v < 8; ++v) {
    int m = mt * 16 + v + 8 * half;  // batch row
    float r = sigf(accI[0][v] + bir + accH[0][v] + bhr);
    float z = sigf(accI[1][v] + biz + accH[1][v] + bhz);
    float n = tanhf(accI[2][v] + bin_ + r * (accH[2][v] + bhn));
    float hp = hcur[(size_t)m * 512 + col];
    float hv = (1.0f - z) * n + z * hp;
    hnxt[(size_t)m * 512 + col] = hv;
    hnxt16[(size_t)m * 512 + col] = (_Float16)hv;
  }
}

// ---------------------------------------------------------------------------
// Fused attention + coverage + context + copy gate. One block per batch row.
// ---------------------------------------------------------------------------
__global__ __launch_bounds__(256) void k_attn(const float* __restrict__ pre,
                                              const float* __restrict__ h,
                                              const float* __restrict__ enc,
                                              const int* __restrict__ src,
                                              const float* __restrict__ W_q,
                                              const float* __restrict__ w_v,
                                              const float* __restrict__ w_cov,
                                              const float* __restrict__ W_copy,
                                              const float* __restrict__ b_copy,
                                              float* __restrict__ ctx,
                                              _Float16* __restrict__ ctx16,
                                              float* __restrict__ cov,
                                              float* __restrict__ copy_pred,
                                              float* __restrict__ attn_last,
                                              float* __restrict__ copy_gate,
                                              float* __restrict__ covloss, int t) {
  __shared__ float hsh[512];
  __shared__ float qk[64];
  __shared__ float attn_sh[256];
  __shared__ float red[256];
  const int b = blockIdx.x, tid = threadIdx.x;
  for (int d = tid; d < 512; d += 256) hsh[d] = h[(size_t)b * 512 + d];
  __syncthreads();
  if (tid < 64) {
    float s = 0.0f;
    #pragma unroll 8
    for (int d = 0; d < 512; ++d) s += hsh[d] * W_q[(size_t)tid * 512 + d];
    qk[tid] = s;
  }
  __syncthreads();
  const int s = tid;  // S == 256 == blockDim
  const float cov_s = cov[b * 256 + s];
  float e = 0.0f;
  #pragma unroll 4
  for (int k = 0; k < 64; ++k)
    e += w_v[k] * tanhf(pre[((size_t)b * 256 + s) * 64 + k] + qk[k] + cov_s * w_cov[k]);
  const float msk = (src[b * 256 + s] == 50256) ? 1.0f : 0.0f;
  e = e * (1.0f - msk) + msk * (-1e6f);
  red[tid] = e; __syncthreads();
  for (int off = 128; off > 0; off >>= 1) {
    if (tid < off) red[tid] = fmaxf(red[tid], red[tid + off]);
    __syncthreads();
  }
  const float mx = red[0]; __syncthreads();
  const float ex = expf(e - mx);
  red[tid] = ex; __syncthreads();
  for (int off = 128; off > 0; off >>= 1) {
    if (tid < off) red[tid] += red[tid + off];
    __syncthreads();
  }
  const float a = ex / red[0];
  attn_sh[s] = a;
  copy_pred[((size_t)b * 64 + t) * 256 + s] = a;
  attn_last[b * 256 + s] = a;
  const float tt = fmaxf((float)t, 1.0f);
  const float cl = fminf(a, cov_s / tt);
  __syncthreads();
  red[tid] = cl; __syncthreads();
  for (int off = 128; off > 0; off >>= 1) {
    if (tid < off) red[tid] += red[tid + off];
    __syncthreads();
  }
  if (tid == 0) covloss[b * 64 + t] = red[0];
  cov[b * 256 + s] = cov_s + a;
  __syncthreads();
  for (int d = tid; d < 512; d += 256) {
    float accd = 0.0f;
    #pragma unroll 4
    for (int ss = 0; ss < 256; ++ss)
      accd += attn_sh[ss] * enc[((size_t)b * 256 + ss) * 512 + d];
    ctx[(size_t)b * 512 + d] = accd;
    ctx16[(size_t)b * 512 + d] = (_Float16)accd;
  }
  __syncthreads();
  float p = 0.0f;
  for (int j = tid; j < 1024; j += 256) {
    float xin = (j < 512) ? hsh[j] : ctx[(size_t)b * 512 + (j - 512)];
    p += xin * W_copy[j];
  }
  red[tid] = p; __syncthreads();
  for (int off = 128; off > 0; off >>= 1) {
    if (tid < off) red[tid] += red[tid + off];
    __syncthreads();
  }
  if (tid == 0) copy_gate[b * 64 + t] = sigf(red[0] + b_copy[0]);
}

// ---------------------------------------------------------------------------
// readout = [emb_t|h|new_ctx] @ W_read^T + b_read (K=1536), double-buffered
// async pipeline, fused maxout via wave32 shfl_xor.
// ---------------------------------------------------------------------------
__global__ __launch_bounds__(256) void k_read(const _Float16* __restrict__ emb16,
                                              const _Float16* __restrict__ h16,
                                              const _Float16* __restrict__ ctx16,
                                              const _Float16* __restrict__ Wread16,
                                              const float* __restrict__ b_read,
                                              float* __restrict__ pred, int t) {
  __shared__ _Float16 lA[2][32 * KP];
  __shared__ _Float16 lB[2][64 * KP];
  const int tid = threadIdx.x, lane = tid & 31, w = tid >> 5;
  const int mt = w & 1, nt = w >> 1;
  const int c0 = blockIdx.x * 64;
  v8f acc = {0, 0, 0, 0, 0, 0, 0, 0};

  auto stage_chunk = [&](int c, int buf) {
    int kk = c * 32;
    if (kk < 512)       stageA(lA[buf], emb16 + (size_t)t * 512 + kk, 64 * 512, tid);
    else if (kk < 1024) stageA(lA[buf], h16 + (kk - 512), 512, tid);
    else                stageA(lA[buf], ctx16 + (kk - 1024), 512, tid);
    stageB(lB[buf], Wread16 + (size_t)c0 * 1536 + kk, 1536, tid);
  };

  stage_chunk(0, 0);
  for (int c = 0; c < 48; ++c) {
    int cur = c & 1;
    if (c + 1 < 48) { stage_chunk(c + 1, cur ^ 1); async_wait<2>(); }
    else            { async_wait<0>(); }
    __syncthreads();
    acc = wmma_step(lA[cur], lB[cur], mt, nt, lane, acc);
    __syncthreads();
  }
  const int col = c0 + nt * 16 + (lane & 15);
  const int half = lane >> 4;
  const float br = b_read[col];
  for (int v = 0; v < 8; ++v) {
    int m = mt * 16 + v + 8 * half;  // batch row
    float val = acc[v] + br;
    float other = __shfl_xor(val, 1, 32);
    float mxo = fmaxf(val, other);
    if ((lane & 1) == 0)
      pred[((size_t)m * 64 + t) * 256 + (col >> 1)] = mxo;
  }
}

// ---------------------------------------------------------------------------
extern "C" void kernel_launch(void* const* d_in, const int* in_sizes, int n_in,
                              void* d_out, int out_size, void* d_ws, size_t ws_size,
                              hipStream_t stream) {
  const int*   tgt    = (const int*)d_in[0];
  const int*   src    = (const int*)d_in[1];
  const float* enc    = (const float*)d_in[2];
  const float* hidden = (const float*)d_in[3];
  const float* wemb   = (const float*)d_in[4];
  const float* W_init = (const float*)d_in[5];
  const float* b_init = (const float*)d_in[6];
  const float* W_ih   = (const float*)d_in[7];
  const float* W_hh   = (const float*)d_in[8];
  const float* b_ih   = (const float*)d_in[9];
  const float* b_hh   = (const float*)d_in[10];
  const float* W_pre  = (const float*)d_in[11];
  const float* b_pre  = (const float*)d_in[12];
  const float* W_q    = (const float*)d_in[13];
  const float* w_v    = (const float*)d_in[14];
  const float* w_cov  = (const float*)d_in[15];
  const float* W_copy = (const float*)d_in[16];
  const float* b_copy = (const float*)d_in[17];
  const float* W_read = (const float*)d_in[18];
  const float* b_read = (const float*)d_in[19];

  float* out       = (float*)d_out;
  float* pred      = out;                 // [32,64,256] = 524288
  float* attn_last = out + 524288;        // [32,256]    = 8192
  float* ctxT      = out + 532480;        // [32,512]    = 16384
  float* copy_pred = out + 548864;        // [32,64,256] = 524288
  float* copy_gate = out + 1073152;       // [32,64]     = 2048
  float* covloss   = out + 1075200;       // [32,64]     = 2048

  float* ws   = (float*)d_ws;
  float* hA   = ws;                 // [32,512]
  float* hB   = hA + 16384;
  float* ctx  = hB + 16384;         // [32,512]
  float* cov  = ctx + 16384;        // [32,256]
  float* adum = cov + 8192;         // attn sink for t<63
  float* pre  = adum + 8192;        // [32,256,64]
  _Float16* f16b    = (_Float16*)(pre + 524288);
  _Float16* hA16    = f16b;                     // [32,512]
  _Float16* hB16    = hA16 + 16384;
  _Float16* ctx16   = hB16 + 16384;             // [32,512]
  _Float16* hid16   = ctx16 + 16384;            // [32,512]
  _Float16* emb16   = hid16 + 16384;            // [32,64,512]
  _Float16* Wpre16  = emb16 + 1048576;          // [64,512]
  _Float16* Winit16 = Wpre16 + 32768;           // [512,512]
  _Float16* Wih16   = Winit16 + 262144;         // [1536,1024]
  _Float16* Whh16   = Wih16 + 1572864;          // [1536,512]
  _Float16* Wread16 = Whh16 + 786432;           // [512,1536]

  hipMemsetAsync(ctx, 0, 16384 * sizeof(float), stream);
  hipMemsetAsync(cov, 0, 8192 * sizeof(float), stream);
  hipMemsetAsync(ctx16, 0, 16384 * sizeof(_Float16), stream);

  k_cvt<<<768, 256, 0, stream>>>(W_ih,   Wih16,   1572864);
  k_cvt<<<384, 256, 0, stream>>>(W_hh,   Whh16,   786432);
  k_cvt<<<384, 256, 0, stream>>>(W_read, Wread16, 786432);
  k_cvt<<<128, 256, 0, stream>>>(W_init, Winit16, 262144);
  k_cvt<<< 16, 256, 0, stream>>>(W_pre,  Wpre16,  32768);
  k_cvt<<<  8, 256, 0, stream>>>(hidden, hid16,   16384);
  k_emb<<<2048, 256, 0, stream>>>(tgt, wemb, emb16);

  k_pre<<<256, 256, 0, stream>>>(enc, Wpre16, b_pre, pre);
  k_init<<<8, 256, 0, stream>>>(hid16, Winit16, b_init, hA, hA16);

  float*    hcur   = hA;   float*    hnxt   = hB;
  _Float16* h16cur = hA16; _Float16* h16nxt = hB16;
  for (int t = 0; t < 64; ++t) {
    k_gru<<<8, 256, 0, stream>>>(emb16, ctx16, h16cur, hcur, Wih16, Whh16,
                                 b_ih, b_hh, hnxt, h16nxt, t);
    k_attn<<<32, 256, 0, stream>>>(pre, hnxt, enc, src, W_q, w_v, w_cov, W_copy, b_copy,
                                   ctx, ctx16, cov, copy_pred,
                                   (t == 63) ? attn_last : adum, copy_gate, covloss, t);
    k_read<<<8, 256, 0, stream>>>(emb16, h16nxt, ctx16, Wread16, b_read, pred, t);
    float* tf = hcur; hcur = hnxt; hnxt = tf;
    _Float16* th = h16cur; h16cur = h16nxt; h16nxt = th;
  }
  hipMemcpyAsync(ctxT, ctx, 16384 * sizeof(float), hipMemcpyDeviceToDevice, stream);
}
